// AnchorHeadWraper_1202590843772
// MI455X (gfx1250) — 8-bit matrix operand order repaired and checked by reading, compiled, not run on hardware
//
#include <hip/hip_runtime.h>
#include <math.h>


#define NUM_CLASSES 80
#define NA 9
#define NB 2
#define TOT 120087
#define NMS_PRE 1000
#define KTOP 256
#define MAXOUT 100
#define SCORE_THR 0.05f
#define IOU_THR 0.5f
#define MAX_RATIO 4.135166556742356f

typedef int v8i __attribute__((ext_vector_type(8)));

__device__ __forceinline__ unsigned int fkey(float x) {
  unsigned int u = __float_as_uint(x);
  return (u & 0x80000000u) ? ~u : (u | 0x80000000u);
}

__global__ void k_zero(unsigned int* p, int n) {
  int t = blockIdx.x * blockDim.x + threadIdx.x;
  if (t < n) p[t] = 0u;
}

__global__ void k_decode(const float* __restrict__ cls, const float* __restrict__ bbox,
                         const float* __restrict__ anc, float* __restrict__ maxs,
                         float* __restrict__ boxes, int HW, int goff) {
  int tid = blockIdx.x * blockDim.x + threadIdx.x;
  int per_img = NA * HW;
  if (tid >= NB * per_img) return;
  int b = tid / per_img;
  int r = tid - b * per_img;
  int a = r / HW;
  int p = r - a * HW;

  const float* cp = cls + ((size_t)(b * NA + a) * NUM_CLASSES) * (size_t)HW + p;
  float m = -1e30f;
#pragma unroll 8
  for (int c = 0; c < NUM_CLASSES; ++c) m = fmaxf(m, cp[(size_t)c * HW]);
  float score = 1.0f / (1.0f + expf(-m));

  const float* bp = bbox + ((size_t)(b * NA + a) * 4) * (size_t)HW + p;
  float d0 = bp[0];
  float d1 = bp[(size_t)HW];
  float d2 = bp[2 * (size_t)HW];
  float d3 = bp[3 * (size_t)HW];

  const float* ap = anc + (size_t)(p * NA + a) * 4;
  float x1 = ap[0], y1 = ap[1], x2 = ap[2], y2 = ap[3];
  float w = x2 - x1, h = y2 - y1;
  float cx = x1 + 0.5f * w, cy = y1 + 0.5f * h;
  float dw = fminf(fmaxf(d2, -MAX_RATIO), MAX_RATIO);
  float dh = fminf(fmaxf(d3, -MAX_RATIO), MAX_RATIO);
  float pcx = cx + d0 * w, pcy = cy + d1 * h;
  float pw = w * expf(dw), ph = h * expf(dh);
  float ox1 = fminf(fmaxf(pcx - 0.5f * pw, 0.f), 800.f);
  float oy1 = fminf(fmaxf(pcy - 0.5f * ph, 0.f), 800.f);
  float ox2 = fminf(fmaxf(pcx + 0.5f * pw, 0.f), 800.f);
  float oy2 = fminf(fmaxf(pcy + 0.5f * ph, 0.f), 800.f);

  int g = b * TOT + goff + (p * NA + a);
  maxs[g] = score;
  float4 bb;
  bb.x = ox1; bb.y = oy1; bb.z = ox2; bb.w = oy2;
  ((float4*)boxes)[g] = bb;
}

__global__ void k_hist1(const float* __restrict__ maxs, unsigned int* hist1) {
  int tid = blockIdx.x * blockDim.x + threadIdx.x;
  if (tid >= NB * TOT) return;
  int b = tid / TOT;
  unsigned int key = fkey(maxs[tid]);
  atomicAdd(&hist1[b * 4096 + (key >> 20)], 1u);
}

__global__ void k_hist2(const float* __restrict__ maxs, const unsigned int* st,
                        unsigned int* hist2) {
  int tid = blockIdx.x * blockDim.x + threadIdx.x;
  if (tid >= NB * TOT) return;
  int b = tid / TOT;
  unsigned int key = fkey(maxs[tid]);
  if ((key >> 20) == st[b * 8 + 0])
    atomicAdd(&hist2[b * 4096 + ((key >> 8) & 0xFFFu)], 1u);
}

__global__ void k_hist3(const float* __restrict__ maxs, const unsigned int* st,
                        unsigned int* hist3) {
  int tid = blockIdx.x * blockDim.x + threadIdx.x;
  if (tid >= NB * TOT) return;
  int b = tid / TOT;
  unsigned int key = fkey(maxs[tid]);
  unsigned int pre = (st[b * 8 + 0] << 12) | st[b * 8 + 1];
  if ((key >> 8) == pre) atomicAdd(&hist3[b * 256 + (key & 0xFFu)], 1u);
}

__global__ void k_scan(const unsigned int* hist1, const unsigned int* hist2,
                       const unsigned int* hist3, unsigned int* st, int pass) {
  int b = blockIdx.x;
  if (threadIdx.x != 0) return;
  unsigned int* s = st + b * 8;
  if (pass == 0) {
    unsigned int acc = 0;
    const unsigned int* h = hist1 + b * 4096;
    for (int i = 4095; i >= 0; --i) {
      acc += h[i];
      if (acc >= NMS_PRE) { s[0] = (unsigned)i; s[6] = acc - h[i]; break; }
    }
  } else if (pass == 1) {
    unsigned int acc = s[6];
    const unsigned int* h = hist2 + b * 4096;
    for (int i = 4095; i >= 0; --i) {
      acc += h[i];
      if (acc >= NMS_PRE) { s[1] = (unsigned)i; s[6] = acc - h[i]; break; }
    }
  } else {
    unsigned int acc = s[6];
    const unsigned int* h = hist3 + b * 256;
    for (int i = 255; i >= 0; --i) {
      acc += h[i];
      if (acc >= NMS_PRE) {
        s[2] = (s[0] << 20) | (s[1] << 8) | (unsigned)i;
        s[3] = NMS_PRE - (acc - h[i]);
        break;
      }
    }
  }
}

__global__ void k_scatter(const float* __restrict__ maxs, unsigned int* st,
                          int* __restrict__ sel_idx) {
  int tid = blockIdx.x * blockDim.x + threadIdx.x;
  if (tid >= NB * TOT) return;
  int b = tid / TOT;
  int n = tid - b * TOT;
  unsigned int key = fkey(maxs[tid]);
  unsigned int* s = st + b * 8;
  unsigned int Kthr = s[2];
  int pos = -1;
  if (key > Kthr) {
    pos = (int)atomicAdd(&s[4], 1u);
  } else if (key == Kthr) {
    unsigned int t = atomicAdd(&s[5], 1u);
    if (t < s[3]) pos = (int)atomicAdd(&s[4], 1u);
  }
  if (pos >= 0 && pos < NMS_PRE) sel_idx[b * NMS_PRE + pos] = n;
}

__global__ void k_gather(const float* c0, const float* c1, const float* c2,
                         const float* c3, const float* c4,
                         const int* __restrict__ sel_idx,
                         const float* __restrict__ boxes,
                         float* __restrict__ s1000, float* __restrict__ b1000) {
  int tid = blockIdx.x * blockDim.x + threadIdx.x;
  if (tid >= NB * NMS_PRE * NUM_CLASSES) return;
  int c = tid % NUM_CLASSES;
  int bi = tid / NUM_CLASSES;
  int b = bi / NMS_PRE;
  int n = sel_idx[bi];
  if (n < 0) n = 0;
  if (n >= TOT) n = TOT - 1;
  int base, HW;
  const float* cp;
  if (n < 90000)       { base = 0;      HW = 10000; cp = c0; }
  else if (n < 112500) { base = 90000;  HW = 2500;  cp = c1; }
  else if (n < 118125) { base = 112500; HW = 625;   cp = c2; }
  else if (n < 119646) { base = 118125; HW = 169;   cp = c3; }
  else                 { base = 119646; HW = 49;    cp = c4; }
  int nl = n - base;
  int a = nl % NA;
  int p = nl / NA;
  float logit = cp[((size_t)(b * NA + a) * NUM_CLASSES + c) * (size_t)HW + p];
  s1000[(size_t)bi * NUM_CLASSES + c] = 1.0f / (1.0f + expf(-logit));
  if (c == 0) ((float4*)b1000)[bi] = ((const float4*)boxes)[b * TOT + n];
}

__global__ __launch_bounds__(256) void k_nms(const float* __restrict__ s1000,
                                             const float* __restrict__ b1000,
                                             float* __restrict__ scores_c,
                                             float* __restrict__ boxes_c) {
  extern __shared__ float smem[];
  float* vals = smem;
  int* idxs = (int*)(smem + 1024);
  float* bx = smem + 2048;
  int* cnt16 = (int*)(smem + 3072);
  unsigned char* keepL = (unsigned char*)(smem + 3088);
  unsigned char* S = (unsigned char*)(smem + 3152);

  int c = blockIdx.x, b = blockIdx.y, tid = threadIdx.x;

  for (int i = tid; i < 1024; i += 256) {
    float v;
    if (i < NMS_PRE) {
      float sc = s1000[((size_t)(b * NMS_PRE) + i) * NUM_CLASSES + c];
      v = sc > SCORE_THR ? sc : -1.0f;
    } else {
      v = -2.0f;
    }
    vals[i] = v;
    idxs[i] = i;
  }
  __syncthreads();

  for (int k = 2; k <= 1024; k <<= 1) {
    for (int j = k >> 1; j > 0; j >>= 1) {
      for (int t = tid; t < 1024; t += 256) {
        int x = t ^ j;
        if (x > t) {
          float va = vals[t], vb = vals[x];
          int ia = idxs[t], ib = idxs[x];
          bool before = (va > vb) || (va == vb && ia < ib);
          bool descSeg = ((t & k) == 0);
          if (before != descSeg) {
            vals[t] = vb; vals[x] = va;
            idxs[t] = ib; idxs[x] = ia;
          }
        }
      }
      __syncthreads();
    }
  }

  {
    int i = tid;
    int id = idxs[i];
    if (id >= NMS_PRE) id = 0;
    float4 bb = ((const float4*)b1000)[b * NMS_PRE + id];
    bx[i * 4 + 0] = bb.x; bx[i * 4 + 1] = bb.y;
    bx[i * 4 + 2] = bb.z; bx[i * 4 + 3] = bb.w;
    keepL[i] = 0;
  }
  __syncthreads();

  {
    int i = tid;
    float ax1 = bx[i * 4], ay1 = bx[i * 4 + 1], ax2 = bx[i * 4 + 2], ay2 = bx[i * 4 + 3];
    float ar = (ax2 - ax1) * (ay2 - ay1);
    for (int j = 0; j < 256; ++j) {
      float px1 = bx[j * 4], py1 = bx[j * 4 + 1], px2 = bx[j * 4 + 2], py2 = bx[j * 4 + 3];
      float br = (px2 - px1) * (py2 - py1);
      float lx = fmaxf(ax1, px1), ly = fmaxf(ay1, py1);
      float rx = fminf(ax2, px2), ry = fminf(ay2, py2);
      float iw = fmaxf(rx - lx, 0.f), ih = fmaxf(ry - ly, 0.f);
      float inter = iw * ih;
      float iou = inter / (ar + br - inter + 1e-6f);
      S[i * 256 + j] = (j < i && iou > IOU_THR) ? (unsigned char)1 : (unsigned char)0;
    }
  }
  __syncthreads();

  {
    int lane = tid;
    int rowm = lane & 15;
    bool hi = lane >= 16;
    bool bl = (lane == 0) || (lane == 16);
    for (int t = 0; t < 16; ++t) {
      if (tid < 32) {
        int i = t * 16 + rowm;
        v8i acc = {0, 0, 0, 0, 0, 0, 0, 0};
#pragma unroll
        for (int kt = 0; kt < 4; ++kt) {
          v8i af, bf;
          const unsigned char* Sr = S + i * 256 + kt * 64;
#pragma unroll
          for (int v = 0; v < 8; ++v) {
            int off = ((v >> 1) << 4) + ((v & 1) << 2) + (hi ? 8 : 0);
            af[v] = *(const int*)(Sr + off);
          }
#pragma unroll
          for (int v = 0; v < 8; ++v) {
            int koff = kt * 64 + ((v >> 1) << 4) + ((v & 1) << 2) + (hi ? 8 : 0);
            int x = *(const int*)(keepL + koff);
            bf[v] = bl ? x : 0;
          }
          acc = __builtin_amdgcn_wmma_i32_16x16x64_iu8(false, af, false, bf, acc,
                                                       false, false);
        }
        if (lane == 0) {
#pragma unroll
          for (int r2 = 0; r2 < 8; ++r2) cnt16[r2] = acc[r2];
        }
        if (lane == 16) {
#pragma unroll
          for (int r2 = 0; r2 < 8; ++r2) cnt16[8 + r2] = acc[r2];
        }
      }
      __syncthreads();
      if (tid == 0) {
        for (int ii = 0; ii < 16; ++ii) {
          int g = t * 16 + ii;
          int sup = (cnt16[ii] > 0) ? 1 : 0;
          for (int jj = 0; jj < ii; ++jj)
            if (keepL[t * 16 + jj] && S[g * 256 + t * 16 + jj]) sup = 1;
          keepL[g] = (!sup && vals[g] > SCORE_THR) ? (unsigned char)1
                                                   : (unsigned char)0;
        }
      }
      __syncthreads();
    }
  }

  {
    int i = tid;
    float outv = keepL[i] ? vals[i] : -1.0f;
    size_t o = ((size_t)(b * NUM_CLASSES) + c) * KTOP + i;
    scores_c[o] = outv;
    float4 ob;
    ob.x = bx[i * 4]; ob.y = bx[i * 4 + 1]; ob.z = bx[i * 4 + 2]; ob.w = bx[i * 4 + 3];
    ((float4*)boxes_c)[o] = ob;
  }
}

__global__ __launch_bounds__(256) void k_final(const float* __restrict__ scores_c,
                                               const float* __restrict__ boxes_c,
                                               float* __restrict__ out) {
  extern __shared__ float fs[];
  float* s = fs;
  float* bv = fs + 20480;
  int* bi = (int*)(fs + 20736);
  int b = blockIdx.x, tid = threadIdx.x;
  for (int f = tid; f < 20480; f += 256) s[f] = scores_c[b * 20480 + f];
  __syncthreads();
  int ndet = 0;
  for (int r = 0; r < MAXOUT; ++r) {
    float mv = -1e30f;
    int mi = 0x7FFFFFFF;
    for (int f = tid; f < 20480; f += 256) {
      float v = s[f];
      if (v > mv || (v == mv && f < mi)) { mv = v; mi = f; }
    }
    bv[tid] = mv; bi[tid] = mi;
    __syncthreads();
    for (int off = 128; off > 0; off >>= 1) {
      if (tid < off) {
        float ov = bv[tid + off];
        int oi = bi[tid + off];
        if (ov > bv[tid] || (ov == bv[tid] && oi < bi[tid])) { bv[tid] = ov; bi[tid] = oi; }
      }
      __syncthreads();
    }
    if (tid == 0) {
      float v = bv[0];
      int f = bi[0];
      bool valid = v > SCORE_THR;
      out[1002 + b * 100 + r] = valid ? (float)(f / KTOP) : -1.0f;
      out[802 + b * 100 + r] = valid ? v : 0.0f;
      float* ob = out + 2 + (size_t)(b * 100 + r) * 4;
      if (valid) {
        float4 bb = ((const float4*)boxes_c)[b * 20480 + f];
        ob[0] = bb.x; ob[1] = bb.y; ob[2] = bb.z; ob[3] = bb.w;
        ndet++;
      } else {
        ob[0] = 0.f; ob[1] = 0.f; ob[2] = 0.f; ob[3] = 0.f;
      }
      s[f] = -1e30f;
    }
    __syncthreads();
  }
  if (tid == 0) out[b] = (float)ndet;
}

extern "C" void kernel_launch(void* const* d_in, const int* in_sizes, int n_in,
                              void* d_out, int out_size, void* d_ws, size_t ws_size,
                              hipStream_t stream) {
  (void)in_sizes; (void)n_in; (void)out_size; (void)ws_size;
  const float* cls[5];
  const float* bbx[5];
  const float* anc[5];
  for (int i = 0; i < 5; ++i) {
    cls[i] = (const float*)d_in[2 * i];
    bbx[i] = (const float*)d_in[2 * i + 1];
    anc[i] = (const float*)d_in[10 + i];
  }
  char* ws = (char*)d_ws;
  size_t off = 0;
  auto alloc = [&](size_t bytes) {
    off = (off + 255) & ~(size_t)255;
    size_t o = off;
    off += bytes;
    return o;
  };
  float* maxs = (float*)(ws + alloc((size_t)NB * TOT * 4));
  float* boxes = (float*)(ws + alloc((size_t)NB * TOT * 16));
  unsigned int* hist1 = (unsigned int*)(ws + alloc((size_t)NB * 4096 * 4));
  unsigned int* hist2 = (unsigned int*)(ws + alloc((size_t)NB * 4096 * 4));
  unsigned int* hist3 = (unsigned int*)(ws + alloc((size_t)NB * 256 * 4));
  unsigned int* st = (unsigned int*)(ws + alloc((size_t)NB * 8 * 4));
  int* sel = (int*)(ws + alloc((size_t)NB * NMS_PRE * 4));
  float* s1000 = (float*)(ws + alloc((size_t)NB * NMS_PRE * NUM_CLASSES * 4));
  float* b1000 = (float*)(ws + alloc((size_t)NB * NMS_PRE * 16));
  float* sc = (float*)(ws + alloc((size_t)NB * NUM_CLASSES * KTOP * 4));
  float* bc = (float*)(ws + alloc((size_t)NB * NUM_CLASSES * KTOP * 16));

  int zn = (int)(((char*)(st + NB * 8)) - (char*)hist1) / 4;
  k_zero<<<(zn + 255) / 256, 256, 0, stream>>>(hist1, zn);

  const int HWs[5] = {10000, 2500, 625, 169, 49};
  const int goffs[5] = {0, 90000, 112500, 118125, 119646};
  for (int l = 0; l < 5; ++l) {
    int total = NB * NA * HWs[l];
    k_decode<<<(total + 255) / 256, 256, 0, stream>>>(cls[l], bbx[l], anc[l], maxs,
                                                      boxes, HWs[l], goffs[l]);
  }
  int tot2 = NB * TOT;
  k_hist1<<<(tot2 + 255) / 256, 256, 0, stream>>>(maxs, hist1);
  k_scan<<<NB, 32, 0, stream>>>(hist1, hist2, hist3, st, 0);
  k_hist2<<<(tot2 + 255) / 256, 256, 0, stream>>>(maxs, st, hist2);
  k_scan<<<NB, 32, 0, stream>>>(hist1, hist2, hist3, st, 1);
  k_hist3<<<(tot2 + 255) / 256, 256, 0, stream>>>(maxs, st, hist3);
  k_scan<<<NB, 32, 0, stream>>>(hist1, hist2, hist3, st, 2);
  k_scatter<<<(tot2 + 255) / 256, 256, 0, stream>>>(maxs, st, sel);

  int gt = NB * NMS_PRE * NUM_CLASSES;
  k_gather<<<(gt + 255) / 256, 256, 0, stream>>>(cls[0], cls[1], cls[2], cls[3], cls[4],
                                                 sel, boxes, s1000, b1000);

  dim3 g(NUM_CLASSES, NB);
  k_nms<<<g, 256, 78144, stream>>>(s1000, b1000, sc, bc);
  k_final<<<NB, 256, 83968, stream>>>(sc, bc, (float*)d_out);
}
